// MLA_25967372272133
// MI455X (gfx1250) — matrix operand order repaired and checked by reading, not compiled, not run on hardware
//
#include <hip/hip_runtime.h>
#include <hip/hip_bf16.h>

typedef __bf16 bf16;
typedef __attribute__((ext_vector_type(16))) __bf16 v16bf;
typedef __attribute__((ext_vector_type(8)))  __bf16 v8bf;
typedef __attribute__((ext_vector_type(4)))  __bf16 v4bf;
typedef __attribute__((ext_vector_type(8)))  float  v8f;
typedef __attribute__((ext_vector_type(4)))  float  v4f;

constexpr int cB = 2, cS = 2048, cDIM = 2048, cH = 16;
constexpr int cQL = 768, cKVL = 512;
constexpr int cNOPE = 128, cROPE = 64, cVD = 128, cQKD = 192;  // QKD = 192
constexpr float cSCALE   = 0.07216878364870322f;  // 192^-0.5
constexpr float cSOFTCAP = 30.0f;
constexpr float cNEG     = -1.0e9f;

// ---------------------------------------------------------------------------
// WMMA fragment loaders (layouts per CDNA5 ISA §7.12.2, wave32)
// A 16x32 bf16: lanes 0-15 row m, K = {k0..k0+7, k0+16..k0+23}
//               lanes 16-31 row m, K = {k0+8..k0+15, k0+24..k0+31}
__device__ __forceinline__ v16bf load_a_frag(const bf16* base, int stride,
                                             int m0, int k0, int lane) {
  int row = m0 + (lane & 15);
  int kc  = k0 + ((lane >> 4) << 3);
  union { v16bf v; v8bf h[2]; } u;
  u.h[0] = *(const v8bf*)(base + (size_t)row * stride + kc);
  u.h[1] = *(const v8bf*)(base + (size_t)row * stride + kc + 16);
  return u.v;
}
// B 32x16 bf16 (KxN), source stored as rows of N each holding K contiguously:
// lane L: N = n0 + L%16, K = {k0+(L/16)*8 .. +7, k0+16+(L/16)*8 .. +7}
__device__ __forceinline__ v16bf load_b_frag(const bf16* base, int stride,
                                             int n0, int k0, int lane) {
  int row = n0 + (lane & 15);
  int kc  = k0 + ((lane >> 4) << 3);
  union { v16bf v; v8bf h[2]; } u;
  u.h[0] = *(const v8bf*)(base + (size_t)row * stride + kc);
  u.h[1] = *(const v8bf*)(base + (size_t)row * stride + kc + 16);
  return u.v;
}

__device__ __forceinline__ v4bf cvt4(v4f f) {
  v4bf h;
  h[0] = (bf16)f[0]; h[1] = (bf16)f[1]; h[2] = (bf16)f[2]; h[3] = (bf16)f[3];
  return h;
}

__device__ __forceinline__ float redmax16(float v) {
#pragma unroll
  for (int m = 1; m < 16; m <<= 1) v = fmaxf(v, __shfl_xor(v, m, 32));
  return v;
}
__device__ __forceinline__ float redsum16(float v) {
#pragma unroll
  for (int m = 1; m < 16; m <<= 1) v += __shfl_xor(v, m, 32);
  return v;
}
__device__ __forceinline__ float redsum32(float v) {
#pragma unroll
  for (int m = 16; m > 0; m >>= 1) v += __shfl_xor(v, m, 32);
  return v;
}

// ---------------------------------------------------------------------------
// Generic GEMM: C[M,N] = A[M,K] * W[N,K]^T   (f32 in, bf16 WMMA, f32 accum)
// Block: 256 threads = 8 waves; tile 128(M) x 128(N), K step 64 (2 wmma steps).
// Wave w: M offset (w&3)*32, N offset (w>>2)*64 -> 16 wmma per barrier pair.
constexpr int GK = 64, GP = 72;  // LDS row stride (pad to kill bank conflicts)

__global__ __launch_bounds__(256) void gemm_bf16_tn(
    const float* __restrict__ A, const float* __restrict__ W,
    float* __restrict__ C, int M, int N, int K) {
  __shared__ bf16 As[128][GP];
  __shared__ bf16 Ws[128][GP];
  const int tid = threadIdx.x, lane = tid & 31, wave = tid >> 5;
  const int m0 = blockIdx.y * 128, n0 = blockIdx.x * 128;
  const int wm = (wave & 3) * 32, wn = (wave >> 2) * 64;

  v8f c[2][4];
#pragma unroll
  for (int i = 0; i < 2; ++i)
#pragma unroll
    for (int j = 0; j < 4; ++j) c[i][j] = (v8f){};

  for (int k0 = 0; k0 < K; k0 += GK) {
    // stage A tile: 128 rows x 64 cols as float4 loads -> packed bf16x4 stores
#pragma unroll
    for (int i = 0; i < 8; ++i) {
      int v = tid + 256 * i;            // vec4 index; 16 vec4 per row
      int r = v >> 4, c4 = (v & 15) * 4;
      v4f f = *(const v4f*)&A[(size_t)(m0 + r) * K + k0 + c4];
      *(v4bf*)&As[r][c4] = cvt4(f);
    }
#pragma unroll
    for (int i = 0; i < 8; ++i) {
      int v = tid + 256 * i;
      int r = v >> 4, c4 = (v & 15) * 4;
      int nr = n0 + r;
      v4f f = (nr < N) ? *(const v4f*)&W[(size_t)nr * K + k0 + c4] : (v4f){};
      *(v4bf*)&Ws[r][c4] = cvt4(f);
    }
    if (k0 + GK < K) {  // hint next K tile toward cache (global_prefetch_b8)
      __builtin_prefetch(&A[(size_t)(m0 + (tid >> 1)) * K + k0 + GK], 0, 0);
    }
    __syncthreads();

#pragma unroll
    for (int kk2 = 0; kk2 < 2; ++kk2) {
      v16bf a0 = load_a_frag(&As[0][0], GP, wm, kk2 * 32, lane);
      v16bf a1 = load_a_frag(&As[0][0], GP, wm + 16, kk2 * 32, lane);
#pragma unroll
      for (int nt = 0; nt < 4; ++nt) {
        v16bf b = load_b_frag(&Ws[0][0], GP, wn + nt * 16, kk2 * 32, lane);
        c[0][nt] = __builtin_amdgcn_wmma_f32_16x16x32_bf16(
            false, a0, false, b, (short)0, c[0][nt], false, false);
        c[1][nt] = __builtin_amdgcn_wmma_f32_16x16x32_bf16(
            false, a1, false, b, (short)0, c[1][nt], false, false);
      }
    }
    __syncthreads();
  }
  // C fragment: VGPR r -> row r (lanes 0-15) / r+8 (lanes 16-31); col = lane%16
#pragma unroll
  for (int mt = 0; mt < 2; ++mt)
#pragma unroll
    for (int nt = 0; nt < 4; ++nt)
#pragma unroll
      for (int r = 0; r < 8; ++r) {
        int m = m0 + wm + mt * 16 + r + ((lane >> 4) << 3);
        int n = n0 + wn + nt * 16 + (lane & 15);
        if (m < M && n < N) C[(size_t)m * N + n] = c[mt][nt][r];
      }
}

// ---------------------------------------------------------------------------
// RMS-norm over rows: y[row, 0:width] = normalize(x[row, 0:width]) * w
__global__ __launch_bounds__(256) void rmsnorm_rows(
    const float* __restrict__ x, const float* __restrict__ w,
    float* __restrict__ y, int rows, int width, int ldx, int ldy) {
  int row = blockIdx.x * (blockDim.x >> 5) + (threadIdx.x >> 5);
  int lane = threadIdx.x & 31;
  if (row >= rows) return;
  const float* xr = x + (size_t)row * ldx;
  float ss = 0.0f;
  for (int i = lane; i < width; i += 32) { float v = xr[i]; ss += v * v; }
  ss = redsum32(ss);
  float r = rsqrtf(ss / (float)width + 1e-6f);
  float* yr = y + (size_t)row * ldy;
  for (int i = lane; i < width; i += 32) yr[i] = xr[i] * r * w[i];
}

// RoPE on k_pe (last 64 cols of kv_full), one block (32 lanes) per token
__global__ __launch_bounds__(32) void rope_kpe_kernel(
    const float* __restrict__ kvfull, const float* __restrict__ cosf,
    const float* __restrict__ sinf, float* __restrict__ kpe) {
  int row = blockIdx.x;        // b*S + s
  int s = row & (cS - 1);
  int i = threadIdx.x;         // pair index 0..31
  const float* src = kvfull + (size_t)row * (cKVL + cROPE) + cKVL;
  float xe = src[2 * i], xo = src[2 * i + 1];
  float c = cosf[s * 32 + i], sn = sinf[s * 32 + i];
  kpe[(size_t)row * cROPE + 2 * i]     = xe * c - xo * sn;
  kpe[(size_t)row * cROPE + 2 * i + 1] = xe * sn + xo * c;
}

// q pack: split nope / rope(q_pe)*gain, emit bf16 head-major (b,h,s,192)
__global__ __launch_bounds__(64) void pack_q_kernel(
    const float* __restrict__ q, const float* __restrict__ cosf,
    const float* __restrict__ sinf, const float* __restrict__ gain,
    bf16* __restrict__ qf) {
  int row = blockIdx.x, h = blockIdx.y, t = threadIdx.x;
  int s = row & (cS - 1), b = row >> 11;
  const float* src = q + (size_t)row * (cH * cQKD) + h * cQKD;
  bf16* dst = qf + ((size_t)(b * cH + h) * cS + s) * cQKD;
  dst[t]      = (bf16)src[t];
  dst[t + 64] = (bf16)src[t + 64];
  if (t < 32) {
    float xe = src[cNOPE + 2 * t], xo = src[cNOPE + 2 * t + 1];
    float c = cosf[s * 32 + t], sn = sinf[s * 32 + t];
    float g = gain[h];
    dst[cNOPE + 2 * t]     = (bf16)((xe * c - xo * sn) * g);
    dst[cNOPE + 2 * t + 1] = (bf16)((xe * sn + xo * c) * g);
  }
}

// k/v pack: k = [k_nope | k_pe broadcast] head-major (b,h,s,192);
// v emitted PRE-TRANSPOSED (b,h,vd,s) so flash stages it with contiguous copies
__global__ __launch_bounds__(64) void pack_kv_kernel(
    const float* __restrict__ kvb, const float* __restrict__ kpe,
    bf16* __restrict__ kk, bf16* __restrict__ vvT) {
  int row = blockIdx.x, h = blockIdx.y, t = threadIdx.x;
  int s = row & (cS - 1), b = row >> 11;
  const float* src = kvb + (size_t)row * (cH * (cNOPE + cVD)) + h * (cNOPE + cVD);
  bf16* kd = kk + ((size_t)(b * cH + h) * cS + s) * cQKD;
  kd[t]         = (bf16)src[t];
  kd[t + 64]    = (bf16)src[t + 64];
  kd[cNOPE + t] = (bf16)kpe[(size_t)row * cROPE + t];
  size_t vbase = (size_t)(b * cH + h) * cVD * cS + s;
  vvT[vbase + (size_t)t * cS]        = (bf16)src[cNOPE + t];
  vvT[vbase + (size_t)(t + 64) * cS] = (bf16)src[cNOPE + 64 + t];
}

// ---------------------------------------------------------------------------
// Flash attention. Grid (S/64, H, B); block = 128 threads = 4 waves.
// Wave w owns 16 query rows; streams 64-key tiles with online softmax.
constexpr int QT = 64, KT2 = 64;
constexpr int KS = 200;  // Ks row stride (192 + pad)
constexpr int VS = 72;   // vT / Ps row stride

__global__ __launch_bounds__(128) void flash_mla_kernel(
    const bf16* __restrict__ qf, const bf16* __restrict__ kk,
    const bf16* __restrict__ vvT, float* __restrict__ y) {
  __shared__ bf16 Ks[KT2][KS];        // key tile, rows = key, cols = 192 feat
  __shared__ bf16 vT[cVD][VS];        // transposed V tile: [vd][key]
  __shared__ bf16 Ps[4][16][VS];      // per-wave softmax probs (16 x 64)

  const int tid = threadIdx.x, lane = tid & 31, wave = tid >> 5;
  const int qt0 = blockIdx.x * QT;
  const int h = blockIdx.y, b = blockIdx.z;
  const size_t bh = (size_t)(b * cH + h);

  // Preload this wave's Q fragments for all 6 K-steps (192 = 6*32)
  const bf16* qbase = qf + (bh * cS + qt0 + wave * 16) * cQKD;
  v16bf qa[6];
#pragma unroll
  for (int ks = 0; ks < 6; ++ks) qa[ks] = load_a_frag(qbase, cQKD, 0, ks * 32, lane);

  v8f o[8];
#pragma unroll
  for (int i = 0; i < 8; ++i) o[i] = (v8f){};
  float rowm[8], rowl[8];
#pragma unroll
  for (int r = 0; r < 8; ++r) { rowm[r] = -INFINITY; rowl[r] = 0.0f; }

  const bf16* kbh = kk + bh * cS * cQKD;
  const bf16* vtbh = vvT + bh * (size_t)cVD * cS;

  for (int j = 0; j <= (int)blockIdx.x; ++j) {  // causal: skip masked key tiles
    const int t0 = j * KT2;
    // stage K tile (64 x 192 bf16), contiguous 16B chunks
#pragma unroll 4
    for (int e = tid; e < KT2 * 24; e += 128) {
      int r = e / 24, ch = (e % 24) * 8;
      *(v8bf*)&Ks[r][ch] = *(const v8bf*)(kbh + (size_t)(t0 + r) * cQKD + ch);
    }
    // stage V^T tile (128 vd-rows x 64 keys), contiguous 16B chunks
#pragma unroll 4
    for (int e = tid; e < cVD * 8; e += 128) {
      int r = e >> 3, ch = (e & 7) * 8;
      *(v8bf*)&vT[r][ch] = *(const v8bf*)(vtbh + (size_t)r * cS + t0 + ch);
    }
    __syncthreads();

    // scores: 16 q-rows x 64 keys = 4 n-tiles, K = 192 (6 steps)
    v8f cs[4];
#pragma unroll
    for (int nt = 0; nt < 4; ++nt) cs[nt] = (v8f){};
#pragma unroll
    for (int ks = 0; ks < 6; ++ks) {
#pragma unroll
      for (int nt = 0; nt < 4; ++nt) {
        v16bf bfrag = load_b_frag(&Ks[0][0], KS, nt * 16, ks * 32, lane);
        cs[nt] = __builtin_amdgcn_wmma_f32_16x16x32_bf16(
            false, qa[ks], false, bfrag, (short)0, cs[nt], false, false);
      }
    }
    // softcap + causal mask + tile row max
    float tmax[8];
#pragma unroll
    for (int r = 0; r < 8; ++r) tmax[r] = -INFINITY;
    const int qrow0 = qt0 + wave * 16 + ((lane >> 4) << 3);
#pragma unroll
    for (int nt = 0; nt < 4; ++nt) {
      int key = t0 + nt * 16 + (lane & 15);
#pragma unroll
      for (int r = 0; r < 8; ++r) {
        float sc = cs[nt][r] * cSCALE;
        sc = cSOFTCAP * tanhf(sc * (1.0f / cSOFTCAP));
        sc += (key > qrow0 + r) ? cNEG : 0.0f;
        cs[nt][r] = sc;
        tmax[r] = fmaxf(tmax[r], sc);
      }
    }
    // online softmax update
    float alpha[8], psum[8];
#pragma unroll
    for (int r = 0; r < 8; ++r) {
      tmax[r] = redmax16(tmax[r]);
      float nm = fmaxf(rowm[r], tmax[r]);
      alpha[r] = __expf(rowm[r] - nm);
      rowm[r] = nm;
      psum[r] = 0.0f;
    }
#pragma unroll
    for (int nt = 0; nt < 4; ++nt)
#pragma unroll
      for (int r = 0; r < 8; ++r) {
        float p = __expf(cs[nt][r] - rowm[r]);
        cs[nt][r] = p;
        psum[r] += p;
      }
#pragma unroll
    for (int r = 0; r < 8; ++r) rowl[r] = rowl[r] * alpha[r] + redsum16(psum[r]);
#pragma unroll
    for (int vt = 0; vt < 8; ++vt)
#pragma unroll
      for (int r = 0; r < 8; ++r) o[vt][r] *= alpha[r];

    // stage P (wave-private LDS region), reload as A fragments for P*V
    bf16* pp = &Ps[wave][0][0];
#pragma unroll
    for (int nt = 0; nt < 4; ++nt)
#pragma unroll
      for (int r = 0; r < 8; ++r)
        pp[(r + ((lane >> 4) << 3)) * VS + nt * 16 + (lane & 15)] = (bf16)cs[nt][r];

#pragma unroll
    for (int ks2 = 0; ks2 < 2; ++ks2) {  // key dim 64 = 2 * 32
      v16bf afrag = load_a_frag(pp, VS, 0, ks2 * 32, lane);
#pragma unroll
      for (int vt = 0; vt < 8; ++vt) {
        v16bf bfrag = load_b_frag(&vT[0][0], VS, vt * 16, ks2 * 32, lane);
        o[vt] = __builtin_amdgcn_wmma_f32_16x16x32_bf16(
            false, afrag, false, bfrag, (short)0, o[vt], false, false);
      }
    }
    __syncthreads();
  }
  // normalize and store y (b, s, h, vd) as f32 rows of H*VD
#pragma unroll
  for (int vt = 0; vt < 8; ++vt)
#pragma unroll
    for (int r = 0; r < 8; ++r) {
      int srow = qt0 + wave * 16 + r + ((lane >> 4) << 3);
      int vd = vt * 16 + (lane & 15);
      y[((size_t)(b * cS + srow)) * (cH * cVD) + h * cVD + vd] = o[vt][r] / rowl[r];
    }
}

// ---------------------------------------------------------------------------
// y <- y - (y . vn) vn, vn = v / max(||v||, eps); one wave per (b,s,h)
__global__ __launch_bounds__(256) void vproj_kernel(
    const float* __restrict__ kvb, float* __restrict__ y, int nrows) {
  int idx = blockIdx.x * (blockDim.x >> 5) + (threadIdx.x >> 5);
  int lane = threadIdx.x & 31;
  if (idx >= nrows) return;
  int h = idx & (cH - 1), row = idx >> 4;  // row = b*S + s
  const float* v = kvb + (size_t)row * (cH * (cNOPE + cVD)) + h * (cNOPE + cVD) + cNOPE;
  float* yr = y + (size_t)row * (cH * cVD) + h * cVD;
  float vl[4], yl[4], ss = 0.0f;
#pragma unroll
  for (int i = 0; i < 4; ++i) { vl[i] = v[lane + 32 * i]; ss += vl[i] * vl[i]; }
  ss = redsum32(ss);
  float rs = rsqrtf(fmaxf(ss, 1e-12f));
  float dot = 0.0f;
#pragma unroll
  for (int i = 0; i < 4; ++i) { yl[i] = yr[lane + 32 * i]; dot += yl[i] * vl[i] * rs; }
  dot = redsum32(dot);
#pragma unroll
  for (int i = 0; i < 4; ++i) yr[lane + 32 * i] = yl[i] - dot * vl[i] * rs;
}

// ---------------------------------------------------------------------------
extern "C" void kernel_launch(void* const* d_in, const int* in_sizes, int n_in,
                              void* d_out, int out_size, void* d_ws, size_t ws_size,
                              hipStream_t stream) {
  const float* x        = (const float*)d_in[0];
  const float* wq_a     = (const float*)d_in[1];
  const float* q_norm_w = (const float*)d_in[2];
  const float* wq_b     = (const float*)d_in[3];
  const float* q_gain   = (const float*)d_in[4];
  const float* wkv_a    = (const float*)d_in[5];
  const float* kv_norm_w= (const float*)d_in[6];
  const float* wkv_b    = (const float*)d_in[7];
  const float* wo       = (const float*)d_in[8];
  const float* cosf     = (const float*)d_in[9];
  const float* sinf     = (const float*)d_in[10];
  // d_in[11] = mask (causal, start_pos=0 -> computed analytically)
  // d_in[12] = start_pos (== 0 in setup)

  const int ROWS = cB * cS;  // 4096
  char* w = (char*)d_ws;
  auto carve = [&](size_t bytes) -> char* {
    char* p = w;
    w += (bytes + 255) & ~(size_t)255;
    return p;
  };
  float* qa     = (float*)carve((size_t)ROWS * cQL * 4);
  float* qn     = (float*)carve((size_t)ROWS * cQL * 4);
  float* qfull  = (float*)carve((size_t)ROWS * cH * cQKD * 4);
  float* kvfull = (float*)carve((size_t)ROWS * (cKVL + cROPE) * 4);
  float* kvn    = (float*)carve((size_t)ROWS * cKVL * 4);
  float* kpe    = (float*)carve((size_t)ROWS * cROPE * 4);
  float* kvb    = (float*)carve((size_t)ROWS * cH * (cNOPE + cVD) * 4);
  bf16*  qfb    = (bf16*)carve((size_t)cB * cH * cS * cQKD * 2);
  bf16*  kkb    = (bf16*)carve((size_t)cB * cH * cS * cQKD * 2);
  bf16*  vvTb   = (bf16*)carve((size_t)cB * cH * cVD * cS * 2);
  float* yacc   = (float*)carve((size_t)ROWS * cH * cVD * 4);

  auto gemm = [&](const float* A, const float* W, float* C, int M, int N, int K) {
    dim3 g((N + 127) / 128, (M + 127) / 128);
    gemm_bf16_tn<<<g, 256, 0, stream>>>(A, W, C, M, N, K);
  };

  // 1) q LoRA down + norm + up
  gemm(x, wq_a, qa, ROWS, cQL, cDIM);
  rmsnorm_rows<<<(ROWS + 7) / 8, 256, 0, stream>>>(qa, q_norm_w, qn, ROWS, cQL, cQL, cQL);
  gemm(qn, wq_b, qfull, ROWS, cH * cQKD, cQL);

  // 2) kv LoRA down + norm + rope(k_pe) + up
  gemm(x, wkv_a, kvfull, ROWS, cKVL + cROPE, cDIM);
  rmsnorm_rows<<<(ROWS + 7) / 8, 256, 0, stream>>>(kvfull, kv_norm_w, kvn, ROWS, cKVL,
                                                   cKVL + cROPE, cKVL);
  rope_kpe_kernel<<<ROWS, 32, 0, stream>>>(kvfull, cosf, sinf, kpe);
  gemm(kvn, wkv_b, kvb, ROWS, cH * (cNOPE + cVD), cKVL);

  // 3) pack bf16 head-major q / k, pre-transposed v
  pack_q_kernel<<<dim3(ROWS, cH), 64, 0, stream>>>(qfull, cosf, sinf, q_gain, qfb);
  pack_kv_kernel<<<dim3(ROWS, cH), 64, 0, stream>>>(kvb, kpe, kkb, vvTb);

  // 4) flash attention (softcap + causal + online softmax)
  flash_mla_kernel<<<dim3(cS / QT, cH, cB), 128, 0, stream>>>(qfb, kkb, vvTb, yacc);

  // 5) remove projection of y onto normalized v
  vproj_kernel<<<(ROWS * cH + 7) / 8, 256, 0, stream>>>(kvb, yacc, ROWS * cH);

  // 6) output projection straight into d_out
  gemm(yacc, wo, (float*)d_out, ROWS, cDIM, cH * cVD);
  (void)in_sizes; (void)n_in; (void)out_size; (void)ws_size;
}
